// DMGI_34127810134592
// MI455X (gfx1250) — matrix operand order repaired and checked by reading, not compiled, not run on hardware
//
#include <hip/hip_runtime.h>
#include <hip/hip_bf16.h>

// Problem constants (from reference): N nodes, F in-feat, H out-feat, R relations, E edges
static constexpr int N_ = 131072;
static constexpr int F_ = 128;
static constexpr int H_ = 64;
static constexpr int R_ = 3;
static constexpr int E_ = 4194304;

typedef __attribute__((ext_vector_type(16))) __bf16 v16bf;
typedef __attribute__((ext_vector_type(8)))  float  v8f;

// ---- bf16 split helpers (round-to-nearest-even) ----
__device__ __forceinline__ unsigned short f32_bf16_rne(float f) {
    unsigned u = __builtin_bit_cast(unsigned, f);
    unsigned r = u + 0x7FFFu + ((u >> 16) & 1u);
    return (unsigned short)(r >> 16);
}
__device__ __forceinline__ float bf16_f32(unsigned short h) {
    unsigned u = ((unsigned)h) << 16;
    return __builtin_bit_cast(float, u);
}
__device__ __forceinline__ __bf16 us_to_bf(unsigned short h) {
    return __builtin_bit_cast(__bf16, h);
}

// =====================================================================
// GEMM: h[N x 64] = x[N x 128] @ W[128 x 64], fp32 in/out.
// Split-bf16: x = xh + xl, W = wh + wl (bf16 each); accumulate
// xh*wh + xh*wl + xl*wh with v_wmma_f32_16x16x32_bf16 (f32 accum).
// Block = 128 threads = 4 waves; each wave owns a 16-row strip,
// computing 4 N-tiles (H=64) over 4 K-steps (F=128) -> 48 WMMAs/wave.
// W is staged in LDS pre-swizzled into B-fragment layout so each lane
// reads its 16 contiguous bf16 (32B) per fragment.
// =====================================================================
__global__ void __launch_bounds__(128)
gemm_xw_wmma(const float* __restrict__ x, const float* __restrict__ Wr,
             float* __restrict__ h)
{
    __shared__ __attribute__((aligned(32))) unsigned short lds_hi[16 * 32 * 16];
    __shared__ __attribute__((aligned(32))) unsigned short lds_lo[16 * 32 * 16];

    const int tid = threadIdx.x;

    // Cooperative convert + swizzle of W into B-fragment layout.
    // Fragment (ik,in0): B is 32x16 (KxN). Lane L: half=L>>4, col=n0+(L&15),
    // element e -> K = k0 + half*16 + e (16 contiguous bf16 per lane).
    for (int idx = tid; idx < 16 * 32; idx += 128) {
        const int frag = idx >> 5;      // 0..15  (= ik*4 + in0)
        const int lane = idx & 31;
        const int ik   = frag >> 2;     // K-step
        const int in0  = frag & 3;      // N-tile
        const int half = lane >> 4;
        const int col  = (in0 << 4) + (lane & 15);
        const int kb   = (ik << 5) + (half << 4);
        unsigned short* dhi = &lds_hi[idx * 16];
        unsigned short* dlo = &lds_lo[idx * 16];
#pragma unroll
        for (int e = 0; e < 16; ++e) {
            float w = Wr[(kb + e) * H_ + col];
            unsigned short hi = f32_bf16_rne(w);
            dhi[e] = hi;
            dlo[e] = f32_bf16_rne(w - bf16_f32(hi));
        }
    }
    __syncthreads();

    const int wave = tid >> 5;
    const int lane = tid & 31;
    const int half = lane >> 4;
    const int m    = lane & 15;
    const int row0 = (blockIdx.x * 4 + wave) * 16;
    const float* xrow = x + (size_t)(row0 + m) * F_;

    v8f acc[4] = {};

#pragma unroll
    for (int ik = 0; ik < 4; ++ik) {
        const int k0 = ik * 32;
        // A fragment 16x32 bf16: lane L (half,m): element e holds
        // K = k0 + half*16 + e  -> one run of 16 floats.
        const int kb = k0 + half * 16;
        float4 q0 = *(const float4*)(xrow + kb);
        float4 q1 = *(const float4*)(xrow + kb + 4);
        float4 q2 = *(const float4*)(xrow + kb + 8);
        float4 q3 = *(const float4*)(xrow + kb + 12);
        float vals[16] = {q0.x, q0.y, q0.z, q0.w, q1.x, q1.y, q1.z, q1.w,
                          q2.x, q2.y, q2.z, q2.w, q3.x, q3.y, q3.z, q3.w};
        v16bf a_hi, a_lo;
#pragma unroll
        for (int e = 0; e < 16; ++e) {
            unsigned short hi = f32_bf16_rne(vals[e]);
            a_hi[e] = us_to_bf(hi);
            a_lo[e] = us_to_bf(f32_bf16_rne(vals[e] - bf16_f32(hi)));
        }
#pragma unroll
        for (int in0 = 0; in0 < 4; ++in0) {
            const int frag = ik * 4 + in0;
            const v16bf b_hi = *(const v16bf*)&lds_hi[(frag * 32 + lane) * 16];
            const v16bf b_lo = *(const v16bf*)&lds_lo[(frag * 32 + lane) * 16];
            acc[in0] = __builtin_amdgcn_wmma_f32_16x16x32_bf16(
                false, a_hi, false, b_hi, (short)0, acc[in0], false, false);
            acc[in0] = __builtin_amdgcn_wmma_f32_16x16x32_bf16(
                false, a_hi, false, b_lo, (short)0, acc[in0], false, false);
            acc[in0] = __builtin_amdgcn_wmma_f32_16x16x32_bf16(
                false, a_lo, false, b_hi, (short)0, acc[in0], false, false);
        }
    }

    // C/D layout: VGPR v, lanes<16 -> row v, lanes>=16 -> row v+8; col = n0+(L&15)
#pragma unroll
    for (int in0 = 0; in0 < 4; ++in0) {
#pragma unroll
        for (int v = 0; v < 8; ++v) {
            const int row = row0 + v + half * 8;
            const int col = in0 * 16 + m;
            h[(size_t)row * H_ + col] = acc[in0][v];
        }
    }
}

// =====================================================================
// Degree / dinv
// =====================================================================
__global__ void __launch_bounds__(256)
deg_kernel(const int* __restrict__ dst, int* __restrict__ deg)
{
    const int e = blockIdx.x * 256 + threadIdx.x;
    atomicAdd(&deg[dst[e]], 1);
}

__global__ void __launch_bounds__(256)
dinv_kernel(const int* __restrict__ deg, float* __restrict__ dinv)
{
    const int i = blockIdx.x * 256 + threadIdx.x;
    dinv[i] = rsqrtf((float)deg[i] + 1.0f);
}

// =====================================================================
// Edge scatter: one wave32 per edge, each lane a float2 column pair.
// agg[dst] += dinv[src]*dinv[dst] * h[ perm? perm[src] : src ]
// h (32MB) is L2-resident; atomics hit L2.
// =====================================================================
__global__ void __launch_bounds__(256)
edge_scatter(const int* __restrict__ src, const int* __restrict__ dst,
             const float* __restrict__ dinv, const float* __restrict__ h,
             const int* __restrict__ perm, float* __restrict__ agg)
{
    const int lane = threadIdx.x & 31;
    const int e    = blockIdx.x * 8 + (threadIdx.x >> 5);
    const int s = src[e];
    const int d = dst[e];
    const float norm = dinv[s] * dinv[d];
    const int j = perm ? perm[s] : s;
    const float2 hv = *(const float2*)(h + (size_t)j * H_ + lane * 2);
    float* a = agg + (size_t)d * H_ + lane * 2;
    atomicAdd(a,     norm * hv.x);
    atomicAdd(a + 1, norm * hv.y);
}

// =====================================================================
// Finalize: out = relu(agg + dinv^2 * h[j] + b); optional column-sum
// reduction (LDS float atomics -> one global atomic per column/block)
// feeding the per-relation summary.
// Block = 256 threads = 4 rows x 64 columns.
// =====================================================================
__global__ void __launch_bounds__(256)
finalize_kernel(const float* __restrict__ h, const float* __restrict__ dinv,
                const float* __restrict__ br, const int* __restrict__ perm,
                float* __restrict__ out, float* __restrict__ sumacc)
{
    __shared__ float lsum[H_];
    const int k = threadIdx.x & 63;
    if (threadIdx.x < H_) lsum[k] = 0.0f;
    __syncthreads();

    const int i = blockIdx.x * 4 + (threadIdx.x >> 6);
    const int j = perm ? perm[i] : i;
    const float di = dinv[i];
    float v = out[(size_t)i * H_ + k] + di * di * h[(size_t)j * H_ + k] + br[k];
    v = fmaxf(v, 0.0f);
    out[(size_t)i * H_ + k] = v;

    if (sumacc) {                    // uniform (kernel-arg) branch
        atomicAdd(&lsum[k], v);      // ds_add_f32
        __syncthreads();
        if (threadIdx.x < H_) atomicAdd(&sumacc[k], lsum[k]);
    }
}

__global__ void summary_kernel(const float* __restrict__ sumacc,
                               float* __restrict__ out_sum)
{
    const int k = threadIdx.x;
    const float m = sumacc[k] * (1.0f / (float)N_);
    out_sum[k] = 1.0f / (1.0f + expf(-m));
}

// =====================================================================
// Host launcher. Inputs: x(N,F) f32, edge_index(R,2,E) i32, perms(R,N) i32,
// W(R,F,H) f32, b(R,H) f32. Output: pos(R,N,H) | neg(R,N,H) | summ(R,1,H).
// Key identity: (x[perm] @ W)[i] = h[perm[i]]  ->  only R GEMMs needed.
// =====================================================================
extern "C" void kernel_launch(void* const* d_in, const int* in_sizes, int n_in,
                              void* d_out, int out_size, void* d_ws, size_t ws_size,
                              hipStream_t stream)
{
    const float* x          = (const float*)d_in[0];
    const int*   edge_index = (const int*)d_in[1];
    const int*   perms      = (const int*)d_in[2];
    const float* W          = (const float*)d_in[3];
    const float* b          = (const float*)d_in[4];
    float*       out        = (float*)d_out;

    // workspace layout
    float* h      = (float*)d_ws;                       // N*H f32 (32 MB, reused per relation)
    int*   deg    = (int*)(h + (size_t)N_ * H_);        // N i32
    float* dinv   = (float*)(deg + N_);                 // N f32
    float* sumacc = dinv + N_;                          // R*H f32

    // zero accumulators (graph-capturable memsets)
    hipMemsetAsync(d_out, 0, (size_t)out_size * sizeof(float), stream);
    hipMemsetAsync(sumacc, 0, (size_t)R_ * H_ * sizeof(float), stream);

    for (int r = 0; r < R_; ++r) {
        const int*   src  = edge_index + (size_t)r * 2 * E_;
        const int*   dst  = src + E_;
        const int*   perm = perms + (size_t)r * N_;
        const float* Wr   = W + (size_t)r * F_ * H_;
        const float* br   = b + (size_t)r * H_;
        float* out_pos = out + (size_t)r * N_ * H_;
        float* out_neg = out + (size_t)(R_ + r) * N_ * H_;
        float* out_sum = out + (size_t)2 * R_ * N_ * H_ + (size_t)r * H_;
        float* sa      = sumacc + r * H_;

        hipMemsetAsync(deg, 0, (size_t)N_ * sizeof(int), stream);
        deg_kernel <<<E_ / 256, 256, 0, stream>>>(dst, deg);
        dinv_kernel<<<N_ / 256, 256, 0, stream>>>(deg, dinv);

        gemm_xw_wmma<<<N_ / 64, 128, 0, stream>>>(x, Wr, h);

        edge_scatter<<<E_ / 8, 256, 0, stream>>>(src, dst, dinv, h, nullptr, out_pos);
        edge_scatter<<<E_ / 8, 256, 0, stream>>>(src, dst, dinv, h, perm,    out_neg);

        finalize_kernel<<<N_ / 4, 256, 0, stream>>>(h, dinv, br, nullptr, out_pos, sa);
        finalize_kernel<<<N_ / 4, 256, 0, stream>>>(h, dinv, br, perm,    out_neg, nullptr);

        summary_kernel<<<1, H_, 0, stream>>>(sa, out_sum);
    }
}
